// VectorQuantizerEMA_30743375905294
// MI455X (gfx1250) — matrix operand order repaired and checked by reading, not compiled, not run on hardware
//
#include <hip/hip_runtime.h>
#include <hip/hip_bf16.h>

// ---------------------------------------------------------------------------
// VQ-VAE vector quantizer for MI455X (gfx1250, wave32, WMMA).
// N = 65536 rows, D = 256, K = 1024 codes.
// d_out layout (f32): [0] vq_loss | [1..1+N*D) quantized_st |
//                     [1+N*D] perplexity | [2+N*D ..) encodings (N x K)
// d_ws layout: [0,4096) u32 hist | [4096,8192) f32 enorm2 |
//              [8192,12288) f32 blockPartial | [16384,540672) bf16 bswz
// ---------------------------------------------------------------------------

typedef __attribute__((ext_vector_type(16))) __bf16 v16bf;
typedef __attribute__((ext_vector_type(8)))  float  v8f;

union BF16x16 { uint4 u[2]; v16bf v; };

#define N_ROWS   65536
#define EMB_D    256
#define KCODES   1024
#define QOFF     ((size_t)1)
#define QELEMS   ((size_t)16777216)           // N_ROWS * EMB_D
#define POFF     ((size_t)16777217)
#define EOFF     ((size_t)16777218)

__device__ __forceinline__ unsigned pack2_bf16(float a, float b) {
    unsigned ua = __float_as_uint(a), ub = __float_as_uint(b);
    ua = (ua + 0x7FFFu + ((ua >> 16) & 1u)) >> 16;   // round-to-nearest-even
    ub = (ub + 0x7FFFu + ((ub >> 16) & 1u)) >> 16;
    return (ua & 0xFFFFu) | (ub << 16);
}

// ---- kernel 1: zero encodings region (8B-aligned base -> float2) + hist ----
__global__ void vq_zero(float2* __restrict__ enc2, unsigned n2,
                        unsigned* __restrict__ hist) {
    unsigned id = blockIdx.x * blockDim.x + threadIdx.x;
    unsigned stride = gridDim.x * blockDim.x;
    if (id < KCODES) hist[id] = 0u;
    float2 z; z.x = 0.0f; z.y = 0.0f;
    for (unsigned v = id; v < n2; v += stride) enc2[v] = z;
}

// ---- kernel 2: exact f32 ||e||^2 per code -------------------------------
__global__ void vq_enorm(const float* __restrict__ emb, float* __restrict__ enorm) {
    int k = blockIdx.x * 256 + threadIdx.x;   // 1024 threads total
    const float4* e4 = (const float4*)(emb + (size_t)k * EMB_D);
    float s = 0.0f;
#pragma unroll 8
    for (int q = 0; q < 64; ++q) {
        float4 v = e4[q];
        s += v.x * v.x + v.y * v.y + v.z * v.z + v.w * v.w;
    }
    enorm[k] = s;
}

// ---- kernel 3: swizzle codebook into WMMA B-operand layout (bf16) --------
// B chunk for code-tile j (16 codes), k-chunk c (32 dims):
//   lane t: col n = t&15, half h = t>>4, element i (0..15) = E[16j+n][32c+16h+i]
// Stored so each (j,c,lane) is 32 contiguous bytes: u32 index == thread id q.
__global__ void vq_bswz(const float* __restrict__ emb, unsigned* __restrict__ bswz) {
    unsigned q = blockIdx.x * blockDim.x + threadIdx.x;   // 131072 threads
    if (q >= 131072u) return;
    int i = (int)(q & 7u) * 2;
    int t = (int)(q >> 3) & 31;
    int c = (int)(q >> 8) & 7;
    int j = (int)(q >> 11);
    int n = t & 15, h = t >> 4;
    const float* src = emb + (size_t)(16 * j + n) * EMB_D + 32 * c + 16 * h + i;
    bswz[q] = pack2_bf16(src[0], src[1]);
}

// ---- kernel 4: main VQ kernel --------------------------------------------
// 128 threads = 4 waves, 64 rows per block; each wave owns 16 rows.
__global__ __launch_bounds__(128) void vq_main(
        const float* __restrict__ inputs, const float* __restrict__ emb,
        const uint4* __restrict__ bswz,  const float* __restrict__ enormWS,
        unsigned* __restrict__ hist, float* __restrict__ blockPartial,
        float* __restrict__ out) {
    __shared__ unsigned short xtile[64 * 264];   // bf16, row stride 528 B (16B pad)
    __shared__ float enorm_s[KCODES];
    __shared__ int   rowArg[64];
    __shared__ float wls[4];

    const int tid  = threadIdx.x;
    const int lane = tid & 31;
    const int wave = tid >> 5;
    const int blockRow = blockIdx.x * 64;

    // Phase 1: coalesced load of 64x256 f32 -> bf16 tile in LDS.
    for (int v = tid; v < 64 * 64; v += 128) {       // 64 float4 per row
        int row = v >> 6, cq = v & 63;
        float4 x = ((const float4*)(inputs + (size_t)(blockRow + row) * EMB_D))[cq];
        unsigned* dst = (unsigned*)((char*)xtile + row * 528 + cq * 8);
        dst[0] = pack2_bf16(x.x, x.y);
        dst[1] = pack2_bf16(x.z, x.w);
    }
    for (int v = tid; v < KCODES; v += 128) enorm_s[v] = enormWS[v];
    __syncthreads();

    // Phase 2: A tile (16 rows x 256) in registers, in this code's contiguous k order (same as bswz B):
    //   lane t: row m = t&15, half h = t>>4; elems 0..15 -> K = 16h+i
    //   (per K-chunk of 32)
    const int n = lane & 15, h = lane >> 4;
    const char* abase = (const char*)xtile + (wave * 16 + n) * 528;
    BF16x16 a[8];
#pragma unroll
    for (int c = 0; c < 8; ++c) {
        a[c].u[0] = *(const uint4*)(abase + 64 * c + 32 * h);
        a[c].u[1] = *(const uint4*)(abase + 64 * c + 32 * h + 16);
    }

    float minv[8];
    int   mini[8];
#pragma unroll
    for (int r = 0; r < 8; ++r) { minv[r] = 3.0e38f; mini[r] = 0; }

    // Sweep 64 code tiles; distance slot r holds (M = r + 8h, code = 16j + n).
    for (int j = 0; j < 64; ++j) {
        if (j < 63)
            __builtin_prefetch((const void*)(bswz + (size_t)(j + 1) * 512 + lane * 2), 0, 1);
        const uint4* bp = bswz + (size_t)j * 512 + lane * 2;
        v8f acc = {0.f, 0.f, 0.f, 0.f, 0.f, 0.f, 0.f, 0.f};
#pragma unroll
        for (int c = 0; c < 8; ++c) {
            BF16x16 b;
            b.u[0] = bp[c * 64 + 0];
            b.u[1] = bp[c * 64 + 1];
            acc = __builtin_amdgcn_wmma_f32_16x16x32_bf16(
                false, a[c].v, false, b.v, (short)0, acc, false, false);
        }
        float en = enorm_s[j * 16 + n];      // exact f32 ||e||^2 (broadcast in half)
        int  code = j * 16 + n;
#pragma unroll
        for (int r = 0; r < 8; ++r) {
            float d = en - 2.0f * acc[r];    // ||x||^2 dropped: row-constant
            if (d < minv[r]) { minv[r] = d; mini[r] = code; }
        }
    }

    // Cross-lane argmin over the 16 lanes of each half (xor masks stay in-half).
#pragma unroll
    for (int r = 0; r < 8; ++r) {
        float v = minv[r]; int ix = mini[r];
#pragma unroll
        for (int off = 1; off < 16; off <<= 1) {
            float ov = __shfl_xor(v, off, 32);
            int   oi = __shfl_xor(ix, off, 32);
            if (ov < v || (ov == v && oi < ix)) { v = ov; ix = oi; }  // jnp tie-break
        }
        if (n == r) rowArg[wave * 16 + h * 8 + r] = ix;
    }
    __syncthreads();

    // Phase 3: gather winning code rows, write quantized (coalesced b32:
    // out+1 is only 4B-aligned), one-hot scatter, histogram, loss partial.
    float lsum = 0.0f;
    for (int m = 0; m < 16; ++m) {
        int row = wave * 16 + m;
        int idx = rowArg[row];
        size_t R = (size_t)(blockRow + row);
        const float* e = emb + (size_t)idx * EMB_D;
        const float* x = inputs + R * EMB_D;
        float* q = out + QOFF + R * EMB_D;
#pragma unroll
        for (int it = 0; it < 8; ++it) {
            int col = lane + it * 32;
            float ev = e[col], xv = x[col];
            q[col] = ev;
            float dv = ev - xv;
            lsum += dv * dv;
        }
        if (lane == 0) {
            atomicAdd(&hist[idx], 1u);
            out[EOFF + R * (size_t)KCODES + (size_t)idx] = 1.0f;
        }
    }
#pragma unroll
    for (int off = 1; off < 32; off <<= 1) lsum += __shfl_xor(lsum, off, 32);
    if (lane == 0) wls[wave] = lsum;
    __syncthreads();
    if (tid == 0)   // deterministic per-block partial (no float atomics)
        blockPartial[blockIdx.x] = wls[0] + wls[1] + wls[2] + wls[3];
}

// ---- kernel 5: finalize loss + perplexity (fixed-order reductions) -------
__global__ void vq_finalize(const unsigned* __restrict__ hist,
                            const float* __restrict__ blockPartial,
                            float* __restrict__ out) {
    __shared__ float redL[256];
    __shared__ float redE[256];
    int tid = threadIdx.x;
    float ls = 0.0f, es = 0.0f;
    for (int k = tid; k < 1024; k += 256) {
        ls += blockPartial[k];
        float p = (float)hist[k] * (1.0f / (float)N_ROWS);
        es += p * logf(p + 1e-10f);
    }
    redL[tid] = ls; redE[tid] = es;
    __syncthreads();
    for (int w = 128; w >= 1; w >>= 1) {
        if (tid < w) { redL[tid] += redL[tid + w]; redE[tid] += redE[tid + w]; }
        __syncthreads();
    }
    if (tid == 0) {
        out[0]    = 0.25f * redL[0] / (float)QELEMS;   // commitment * mean sq err
        out[POFF] = expf(-redE[0]);
    }
}

extern "C" void kernel_launch(void* const* d_in, const int* in_sizes, int n_in,
                              void* d_out, int out_size, void* d_ws, size_t ws_size,
                              hipStream_t stream) {
    const float* inputs = (const float*)d_in[0];   // [65536, 256] f32
    const float* emb    = (const float*)d_in[1];   // [1024, 256]  f32
    float* out = (float*)d_out;

    char* ws = (char*)d_ws;                        // needs ~528 KB
    unsigned* hist   = (unsigned*)ws;              // [0, 4096)
    float* enorm     = (float*)(ws + 4096);        // [4096, 8192)
    float* bpart     = (float*)(ws + 8192);        // [8192, 12288)
    uint4* bswz      = (uint4*)(ws + 16384);       // [16384, 540672)

    // encodings region: 67108864 f32 = 33554432 float2 (base is 8B aligned)
    vq_zero<<<2048, 256, 0, stream>>>((float2*)(out + EOFF), 33554432u, hist);
    vq_enorm<<<4, 256, 0, stream>>>(emb, enorm);
    vq_bswz<<<512, 256, 0, stream>>>(emb, (unsigned*)bswz);
    vq_main<<<1024, 128, 0, stream>>>(inputs, emb, bswz, enorm, hist, bpart, out);
    vq_finalize<<<1, 256, 0, stream>>>(hist, bpart, out);
}
